// AVNNType1Linear_19902878450221
// MI455X (gfx1250) — matrix operand order repaired and checked by reading, not compiled, not run on hardware
//
#include <hip/hip_runtime.h>
#include <hip/hip_bf16.h>

typedef unsigned short u16;
typedef __attribute__((ext_vector_type(16))) __bf16 v16bf;
typedef __attribute__((ext_vector_type(8)))  float  v8f;
typedef __attribute__((ext_vector_type(4)))  float  f32x4;
typedef __attribute__((ext_vector_type(2)))  float  f32x2;
typedef int b128v __attribute__((vector_size(16)));   // 4 x i32

union FragAB { v16bf v; uint4 q[2]; };

#if __has_builtin(__builtin_amdgcn_global_load_async_to_lds_b128) && \
    __has_builtin(__builtin_amdgcn_s_wait_asynccnt)
#define HAVE_ASYNC_LDS 1
#else
#define HAVE_ASYNC_LDS 0
#endif

__device__ __forceinline__ void async_copy_b128(const void* g, void* l) {
#if HAVE_ASYNC_LDS
    __builtin_amdgcn_global_load_async_to_lds_b128(
        (__attribute__((address_space(1))) b128v*)g,
        (__attribute__((address_space(3))) b128v*)l,
        /*offset=*/0, /*cpol=*/0);
#else
    (void)g; (void)l;
#endif
}

__device__ __forceinline__ u16 f32_to_bf16(float f) {
    unsigned int u = __float_as_uint(f);
    unsigned int r = u + 0x7FFFu + ((u >> 16) & 1u);   // round-to-nearest-even
    return (u16)(r >> 16);
}

// ---------------------------------------------------------------------------
// Kernel 1: W fp32 [D_OUT, D_IN] -> bf16 workspace (streaming, read-once)
// ---------------------------------------------------------------------------
__global__ void convert_W_kernel(const float* __restrict__ W,
                                 u16* __restrict__ Wb, long total4) {
    long idx4 = (long)blockIdx.x * blockDim.x + threadIdx.x;
    if (idx4 >= total4) return;
    f32x4 f = __builtin_nontemporal_load(reinterpret_cast<const f32x4*>(W) + idx4);
    union { u16 h[4]; uint2 u; } p;
    p.h[0] = f32_to_bf16(f.x);
    p.h[1] = f32_to_bf16(f.y);
    p.h[2] = f32_to_bf16(f.z);
    p.h[3] = f32_to_bf16(f.w);
    *reinterpret_cast<uint2*>(Wb + idx4 * 4) = p.u;
}

// ---------------------------------------------------------------------------
// Kernel 2: de-interleave activator channel -> bf16, reduce carry scalar.
// ---------------------------------------------------------------------------
__global__ void prep_x_kernel(const float* __restrict__ x,   // [B, D_IN, 2]
                              u16* __restrict__ Ab,          // [B, D_IN] bf16
                              float* __restrict__ val,       // [B]
                              int D_IN) {
    const int b = blockIdx.x;
    const int t = threadIdx.x;
    const f32x2* xr = reinterpret_cast<const f32x2*>(x) + (size_t)b * D_IN;
    u16* ar = Ab + (size_t)b * D_IN;

    float s = 0.f;
    for (int i = t; i < D_IN; i += 256) {
        f32x2 v = __builtin_nontemporal_load(xr + i);  // .x = act, .y = carry
        ar[i] = f32_to_bf16(v.x);
        s += v.x + v.y;
    }
    __shared__ float red[256];
    red[t] = s;
    __syncthreads();
    for (int off = 128; off > 0; off >>= 1) {
        if (t < off) red[t] += red[t + off];
        __syncthreads();
    }
    if (t == 0) val[b] = red[0] * (0.5f / (float)D_IN);
}

// ---------------------------------------------------------------------------
// Kernel 3: bf16 WMMA GEMM + fused bias/ReLU/carry-interleave epilogue.
// Block = 256 threads (8 waves). Block tile: 128 (M) x 128 (N), K-step 32.
// Wave w: M = (w&3)*32 (2 sub-tiles), N = (w>>2)*64 (4 sub-tiles) -> 8 WMMAs
// per K-step from 2 A + 4 B fragments. Double-buffered LDS, 1 barrier/K-step.
// Tiles staged with GLOBAL_LOAD_ASYNC_TO_LDS_B128 (ASYNCcnt) when available,
// issued before the WMMAs so the DMA overlaps the whole compute phase.
// Rows padded to 40 elements (80 B = 20-bank stride): conflict-free frags.
// ---------------------------------------------------------------------------
#define SLD 40   // padded LDS row stride in bf16 elements (32 data + 8 pad)

__global__ void __launch_bounds__(256)
gemm_bf16_kernel(const u16* __restrict__ A,     // [B, K] bf16
                 const u16* __restrict__ Wb,    // [N, K] bf16
                 const float* __restrict__ bias,// [N]
                 const float* __restrict__ val, // [B]
                 float* __restrict__ out,       // [B, N, 2]
                 int K, int N) {
    __shared__ __align__(16) u16 sA[2][128 * SLD];
    __shared__ __align__(16) u16 sB[2][128 * SLD];

    const int t     = threadIdx.x;
    const int lane  = t & 31;
    const int wave  = t >> 5;
    const int mBase = blockIdx.y * 128;
    const int nBase = blockIdx.x * 128;
    const int mg    = (wave & 3) * 32;    // wave's M sub-range (2 x 16)
    const int ng    = (wave >> 2) * 64;   // wave's N sub-range (4 x 16)

    // Cooperative load mapping: tile 128x32 = 512 x (8 bf16 chunks);
    // thread handles chunks t and t+256: row = e/4, col = (e%4)*8.
    const int r0 = t >> 2,          c0 = (t & 3) * 8;
    const int r1 = (t + 256) >> 2,  c1 = c0;

    const u16* gA0 = A  + (size_t)(mBase + r0) * K + c0;
    const u16* gA1 = A  + (size_t)(mBase + r1) * K + c1;
    const u16* gB0 = Wb + (size_t)(nBase + r0) * K + c0;
    const u16* gB1 = Wb + (size_t)(nBase + r1) * K + c1;

    const int nK = K / 32;

#if HAVE_ASYNC_LDS
    // ---- async DMA tile 0 ----
    async_copy_b128(gA0, &sA[0][r0 * SLD + c0]);
    async_copy_b128(gA1, &sA[0][r1 * SLD + c1]);
    async_copy_b128(gB0, &sB[0][r0 * SLD + c0]);
    async_copy_b128(gB1, &sB[0][r1 * SLD + c1]);
    __builtin_amdgcn_s_wait_asynccnt(0);
    __syncthreads();
#else
    uint4 pa0 = *reinterpret_cast<const uint4*>(gA0);
    uint4 pa1 = *reinterpret_cast<const uint4*>(gA1);
    uint4 pb0 = *reinterpret_cast<const uint4*>(gB0);
    uint4 pb1 = *reinterpret_cast<const uint4*>(gB1);
    *reinterpret_cast<uint4*>(&sA[0][r0 * SLD + c0]) = pa0;
    *reinterpret_cast<uint4*>(&sA[0][r1 * SLD + c1]) = pa1;
    *reinterpret_cast<uint4*>(&sB[0][r0 * SLD + c0]) = pb0;
    *reinterpret_cast<uint4*>(&sB[0][r1 * SLD + c1]) = pb1;
    __syncthreads();
#endif

    v8f acc[2][4] = {};

    // Fragment addressing (documented VGPR layouts):
    //  A 16x32 bf16: lane<16 -> K 0..7 (q0) & 16..23 (q1); lane>=16 -> K 8..15 & 24..31
    //  B 32x16 bf16: lane<16 -> K 0..7 (q0) & 16..23 (q1); lane>=16 -> K 8..15 & 24..31; lane%16 = N
    const int l15 = lane & 15;
    const int aKb = (lane < 16) ? 0 : 8;
    const int bKb = (lane < 16) ? 0 : 8;

    int cur = 0;
    for (int kt = 0; kt < nK; ++kt) {
        const bool more = (kt + 1 < nK);
        const int  k0   = (kt + 1) * 32;
        const int  nxt  = cur ^ 1;

#if HAVE_ASYNC_LDS
        if (more) {   // DMA next tile while we compute on the current one
            async_copy_b128(gA0 + k0, &sA[nxt][r0 * SLD + c0]);
            async_copy_b128(gA1 + k0, &sA[nxt][r1 * SLD + c1]);
            async_copy_b128(gB0 + k0, &sB[nxt][r0 * SLD + c0]);
            async_copy_b128(gB1 + k0, &sB[nxt][r1 * SLD + c1]);
        }
#else
        uint4 na0, na1, nb0, nb1;
        if (more) {
            na0 = *reinterpret_cast<const uint4*>(gA0 + k0);
            na1 = *reinterpret_cast<const uint4*>(gA1 + k0);
            nb0 = *reinterpret_cast<const uint4*>(gB0 + k0);
            nb1 = *reinterpret_cast<const uint4*>(gB1 + k0);
        }
#endif

        // load ALL fragments first so the 8 WMMAs can issue back-to-back
        const u16* __restrict__ sa = &sA[cur][0];
        const u16* __restrict__ sb = &sB[cur][0];
        FragAB fa[2], fb[4];
        #pragma unroll
        for (int mi = 0; mi < 2; ++mi) {
            const int row = mg + mi * 16 + l15;
            fa[mi].q[0] = *reinterpret_cast<const uint4*>(&sa[row * SLD + aKb]);
            fa[mi].q[1] = *reinterpret_cast<const uint4*>(&sa[row * SLD + aKb + 16]);
        }
        #pragma unroll
        for (int ni = 0; ni < 4; ++ni) {
            const int row = ng + ni * 16 + l15;
            fb[ni].q[0] = *reinterpret_cast<const uint4*>(&sb[row * SLD + bKb]);
            fb[ni].q[1] = *reinterpret_cast<const uint4*>(&sb[row * SLD + bKb + 16]);
        }

        #pragma unroll
        for (int mi = 0; mi < 2; ++mi)
            #pragma unroll
            for (int ni = 0; ni < 4; ++ni)
                acc[mi][ni] = __builtin_amdgcn_wmma_f32_16x16x32_bf16(
                    false, fa[mi].v, false, fb[ni].v,
                    (short)0, acc[mi][ni], false, false);

#if HAVE_ASYNC_LDS
        if (more) __builtin_amdgcn_s_wait_asynccnt(0);
#else
        if (more) {
            *reinterpret_cast<uint4*>(&sA[nxt][r0 * SLD + c0]) = na0;
            *reinterpret_cast<uint4*>(&sA[nxt][r1 * SLD + c1]) = na1;
            *reinterpret_cast<uint4*>(&sB[nxt][r0 * SLD + c0]) = nb0;
            *reinterpret_cast<uint4*>(&sB[nxt][r1 * SLD + c1]) = nb1;
        }
#endif
        __syncthreads();
        cur ^= 1;
    }

    // ---- epilogue ----
    // C/D layout: VGPR r -> M = r (lanes 0-15) or r+8 (lanes 16-31), N = lane%16
    const int mHalf = (lane >> 4) << 3;
    float cvv[2][8];
    #pragma unroll
    for (int mi = 0; mi < 2; ++mi)
        #pragma unroll
        for (int r = 0; r < 8; ++r)
            cvv[mi][r] = val[mBase + mg + mi * 16 + r + mHalf];

    #pragma unroll
    for (int ni = 0; ni < 4; ++ni) {
        const int nGlob = nBase + ng + ni * 16 + l15;
        const float bv = bias[nGlob];
        #pragma unroll
        for (int mi = 0; mi < 2; ++mi) {
            #pragma unroll
            for (int r = 0; r < 8; ++r) {
                const int mGlob = mBase + mg + mi * 16 + r + mHalf;
                float a = acc[mi][ni][r] + bv;
                a = fmaxf(a, 0.f);
                f32x2 o2; o2.x = a; o2.y = cvv[mi][r];
                __builtin_nontemporal_store(
                    o2, reinterpret_cast<f32x2*>(out + ((size_t)mGlob * N + nGlob) * 2));
            }
        }
    }
}

// ---------------------------------------------------------------------------
extern "C" void kernel_launch(void* const* d_in, const int* in_sizes, int n_in,
                              void* d_out, int out_size, void* d_ws, size_t ws_size,
                              hipStream_t stream) {
    const float* x  = (const float*)d_in[0];   // [B, D_IN, 2]
    const float* W  = (const float*)d_in[1];   // [D_OUT, D_IN]
    const float* b  = (const float*)d_in[2];   // [D_OUT]
    float* out = (float*)d_out;                // [B, D_OUT, 2]

    const int  D_OUT = in_sizes[2];
    const long D_IN  = in_sizes[1] / D_OUT;
    const long B     = in_sizes[0] / (2 * D_IN);

    // workspace layout: W_bf16 | A_bf16 | val
    u16*   Wb  = (u16*)d_ws;
    u16*   Ab  = Wb + (size_t)D_OUT * D_IN;
    float* val = (float*)(Ab + (size_t)B * D_IN);

    {   // W fp32 -> bf16
        long total4 = ((long)D_OUT * D_IN) / 4;
        long blocks = (total4 + 255) / 256;
        convert_W_kernel<<<dim3((unsigned)blocks), dim3(256), 0, stream>>>(W, Wb, total4);
    }
    {   // de-interleave + per-row carry scalar
        prep_x_kernel<<<dim3((unsigned)B), dim3(256), 0, stream>>>(x, Ab, val, (int)D_IN);
    }
    {   // WMMA GEMM + fused epilogue
        dim3 grid((unsigned)(D_OUT / 128), (unsigned)(B / 128));
        gemm_bf16_kernel<<<grid, dim3(256), 0, stream>>>(Ab, Wb, b, val, out,
                                                         (int)D_IN, D_OUT);
    }
}
